// QRNNDecoder_48404281426008
// MI455X (gfx1250) — compile-verified, matrix operand order checked by reading, not run on hardware
//
#include <hip/hip_runtime.h>
#include <hip/hip_bf16.h>
#include <math.h>

// ---- problem constants ----
#define BB   32
#define TT   64
#define SS   128
#define EMB  512
#define HH   512
#define EE   512
#define VV   32000
#define LDK  40   // padded LDS K-stride (bf16 elems): 80B rows keep 16B alignment

typedef __attribute__((ext_vector_type(16))) __bf16 v16bf;
typedef __attribute__((ext_vector_type(8)))  __bf16 v8bf;
typedef __attribute__((ext_vector_type(8)))  float  v8f;

struct FragBF {
    union {
        v16bf v;
        struct { v8bf lo, hi; } h;
    };
};

// Load one 16(row) x 32(K) bf16 fragment from a row-major tile (LDS or global).
// ISA layout: lane&15 = row, lane&16 selects K-half; two 16B chunks per lane.
__device__ __forceinline__ v16bf ld_frag(const __bf16* base, int ldk) {
    int lane = threadIdx.x & 31;
    const __bf16* p = base + (lane & 15) * ldk + ((lane & 16) ? 8 : 0);
    FragBF f;
    f.h.lo = *reinterpret_cast<const v8bf*>(p);
    f.h.hi = *reinterpret_cast<const v8bf*>(p + 16);
    return f.v;
}

__device__ __forceinline__ v8f wmma_bf16(v16bf a, v16bf b, v8f c) {
    return __builtin_amdgcn_wmma_f32_16x16x32_bf16(
        false, a, false, b, (short)0, c, false, false);
}

__device__ __forceinline__ float sigmoidf_(float x) {
    return 1.0f / (1.0f + expf(-x));
}

// 8 fp32 -> packed v8bf (two 16B global reads, one 16B LDS store at call site)
__device__ __forceinline__ v8bf cvt8(const float* __restrict__ src) {
    float4 f0 = *reinterpret_cast<const float4*>(src);
    float4 f1 = *reinterpret_cast<const float4*>(src + 4);
    v8bf o;
    o[0] = (__bf16)f0.x; o[1] = (__bf16)f0.y; o[2] = (__bf16)f0.z; o[3] = (__bf16)f0.w;
    o[4] = (__bf16)f1.x; o[5] = (__bf16)f1.y; o[6] = (__bf16)f1.z; o[7] = (__bf16)f1.w;
    return o;
}

// ---- CDNA5 async LDS-DMA (ASYNCcnt) ----
typedef __attribute__((address_space(3))) const void* lds_cvptr;
__device__ __forceinline__ uint32_t lds_off(const void* p) {
    return (uint32_t)(uintptr_t)(lds_cvptr)p;
}
__device__ __forceinline__ void async_ld_b128(uint32_t lds_addr, const void* gaddr) {
    asm volatile("global_load_async_to_lds_b128 %0, %1, off"
                 :: "v"(lds_addr), "v"(gaddr) : "memory");
}
__device__ __forceinline__ void wait_async0() {
    asm volatile("s_wait_asynccnt 0" ::: "memory");
}

// ---------------------------------------------------------------------------
// Repack one gate weight tensor (H,EMB,2) fp32 -> bf16 [tap][n][k] contiguous
// ---------------------------------------------------------------------------
__global__ __launch_bounds__(256) void k_repack_tap(
    const float* __restrict__ W, __bf16* __restrict__ dst)
{
    int idx = blockIdx.x * 256 + threadIdx.x;   // over 512*512*2
    if (idx >= 512 * 1024) return;
    int n = idx >> 10, rem = idx & 1023, k = rem >> 1, tap = rem & 1;
    dst[((size_t)tap * 512 + n) * 512 + k] = (__bf16)W[idx];
}

__global__ __launch_bounds__(256) void k_cvt(
    const float* __restrict__ src, __bf16* __restrict__ dst, int n)
{
    int idx = blockIdx.x * 256 + threadIdx.x;
    if (idx < n) dst[idx] = (__bf16)src[idx];
}

// ---------------------------------------------------------------------------
// ep[w][b][n] = encoder_hidden[w<3?0:1][b] . W_w[n,:] + bias_w[n]
// ---------------------------------------------------------------------------
__global__ __launch_bounds__(256) void k_prep_ep(
    const float* __restrict__ hid,
    const float* W0, const float* b0, const float* W1, const float* b1,
    const float* W2, const float* b2, const float* W3, const float* b3,
    const float* W4, const float* b4, float* __restrict__ ep)
{
    int gidx = blockIdx.x * 256 + threadIdx.x;
    if (gidx >= 5 * BB * HH) return;
    int w = gidx / (BB * HH);
    int rem = gidx - w * (BB * HH);
    int b = rem >> 9;
    int n = rem & 511;
    const float* Ws[5] = {W0, W1, W2, W3, W4};
    const float* bs[5] = {b0, b1, b2, b3, b4};
    const float* h = hid + ((w < 3) ? 0 : BB * EE) + (size_t)b * EE;
    const float* wr = Ws[w] + (size_t)n * EE;
    float acc = 0.f;
    for (int e = 0; e < EE; ++e) acc += h[e] * wr[e];
    ep[gidx] = acc + bs[w][n];
}

__global__ void k_zero(unsigned int* p, int n) {
    int i = blockIdx.x * blockDim.x + threadIdx.x;
    if (i < n) p[i] = 0u;
}

// ---------------------------------------------------------------------------
// Layer 0 gates: 8 blocks x 128 threads; wave g in {z,f,o,i}.
// B fragments come straight from repacked bf16 weights in global (L2-hot).
// ---------------------------------------------------------------------------
__global__ __launch_bounds__(128) void k_layer0(
    const float* __restrict__ embed, int t,
    const __bf16* __restrict__ w0bf,
    const float* zb, const float* fb, const float* ob, const float* ib,
    const float* __restrict__ ep, float* __restrict__ c0,
    __bf16* __restrict__ h0out)
{
    __shared__ __bf16 sX[32 * LDK];
    __shared__ float  sG[4 * 32 * 64];
    const int tid = threadIdx.x, wv = tid >> 5, lane = tid & 31;
    const int n0 = blockIdx.x * 64;

    v8f acc[2][4];
    for (int a = 0; a < 2; ++a)
        for (int b = 0; b < 4; ++b) acc[a][b] = {};

    for (int tap = 0; tap < 2; ++tap) {
        const int tsel = (tap == 0) ? (t - 1) : t;
        for (int kc = 0; kc < 16; ++kc) {
            __syncthreads();
            {   // stage X tile (32x32, fp32->bf16): one 8-wide item per thread
                int r = tid >> 2, c8 = (tid & 3) * 8;
                v8bf o = {};
                if (tsel >= 0)
                    o = cvt8(&embed[((size_t)r * TT + tsel) * EMB + kc * 32 + c8]);
                *reinterpret_cast<v8bf*>(&sX[r * LDK + c8]) = o;
            }
            __syncthreads();
            v16bf a0 = ld_frag(sX, LDK);
            v16bf a1 = ld_frag(sX + 16 * LDK, LDK);
            const __bf16* wb =
                w0bf + ((size_t)(wv * 2 + tap) * HH + n0) * 512 + kc * 32;
            for (int nt = 0; nt < 4; ++nt) {
                v16bf bfr = ld_frag(wb + nt * 16 * 512, 512);
                acc[0][nt] = wmma_bf16(a0, bfr, acc[0][nt]);
                acc[1][nt] = wmma_bf16(a1, bfr, acc[1][nt]);
            }
        }
    }
    {   // gate activation epilogue -> LDS
        const int g = wv;
        const float* bias = (g == 0) ? zb : (g == 1) ? fb : (g == 2) ? ob : ib;
        const float* epg  = ep + ((g <= 1) ? 0 : (g == 2) ? 1 : 2) * BB * HH;
        const int nl = lane & 15, mh = (lane & 16) ? 8 : 0;
        for (int mt = 0; mt < 2; ++mt)
            for (int nt = 0; nt < 4; ++nt) {
                v8f a = acc[mt][nt];
                for (int i = 0; i < 8; ++i) {
                    int m = mt * 16 + mh + i;
                    int nn = nt * 16 + nl;
                    float v = a[i] + bias[n0 + nn] + epg[m * HH + n0 + nn];
                    v = (g == 0) ? tanhf(v) : sigmoidf_(v);
                    sG[(g * 32 + m) * 64 + nn] = v;
                }
            }
    }
    __syncthreads();
    // c0 = f*c0 + i*z ; h0 = o*c0
    for (int l = tid; l < 32 * 64; l += 128) {
        int m = l >> 6, nn = l & 63;
        float z  = sG[(0 * 32 + m) * 64 + nn];
        float f  = sG[(1 * 32 + m) * 64 + nn];
        float o  = sG[(2 * 32 + m) * 64 + nn];
        float ii = sG[(3 * 32 + m) * 64 + nn];
        size_t idx = (size_t)m * HH + n0 + nn;
        float c = f * c0[idx] + ii * z;
        c0[idx] = c;
        h0out[idx] = (__bf16)(o * c);
    }
}

// ---------------------------------------------------------------------------
// Layer 1 gates (3 gates): fully LDS-staging-free; A and B frags from global.
// ---------------------------------------------------------------------------
__global__ __launch_bounds__(128) void k_layer1(
    const __bf16* __restrict__ h0prev, const __bf16* __restrict__ h0cur,
    const __bf16* __restrict__ w1bf,
    const float* zb, const float* fb, const float* ob,
    const float* __restrict__ ep, const float* __restrict__ h1old,
    float* __restrict__ c1nf, float* __restrict__ o1f)
{
    __shared__ float sG[3 * 32 * 64];
    const int tid = threadIdx.x, wv = tid >> 5, lane = tid & 31;
    const int n0 = blockIdx.x * 64;

    if (wv < 3) {
        v8f acc[2][4];
        for (int a = 0; a < 2; ++a)
            for (int b = 0; b < 4; ++b) acc[a][b] = {};

        for (int tap = 0; tap < 2; ++tap) {
            const __bf16* hsrc = tap ? h0cur : h0prev;
            for (int kc = 0; kc < 16; ++kc) {
                v16bf a0 = ld_frag(hsrc + kc * 32, HH);
                v16bf a1 = ld_frag(hsrc + (size_t)16 * HH + kc * 32, HH);
                const __bf16* wb =
                    w1bf + ((size_t)(wv * 2 + tap) * HH + n0) * 512 + kc * 32;
                for (int nt = 0; nt < 4; ++nt) {
                    v16bf bfr = ld_frag(wb + nt * 16 * 512, 512);
                    acc[0][nt] = wmma_bf16(a0, bfr, acc[0][nt]);
                    acc[1][nt] = wmma_bf16(a1, bfr, acc[1][nt]);
                }
            }
        }
        const int g = wv;
        const float* bias = (g == 0) ? zb : (g == 1) ? fb : ob;
        const float* epg  = ep + ((g <= 1) ? 3 : 4) * BB * HH;
        const int nl = lane & 15, mh = (lane & 16) ? 8 : 0;
        for (int mt = 0; mt < 2; ++mt)
            for (int nt = 0; nt < 4; ++nt) {
                v8f a = acc[mt][nt];
                for (int i = 0; i < 8; ++i) {
                    int m = mt * 16 + mh + i;
                    int nn = nt * 16 + nl;
                    float v = a[i] + bias[n0 + nn] + epg[m * HH + n0 + nn];
                    v = (g == 0) ? tanhf(v) : sigmoidf_(v);
                    sG[(g * 32 + m) * 64 + nn] = v;
                }
            }
    }
    __syncthreads();
    // c1n = f*h1_old + (1-f)*z ; store o1
    for (int l = tid; l < 32 * 64; l += 128) {
        int m = l >> 6, nn = l & 63;
        float z = sG[(0 * 32 + m) * 64 + nn];
        float f = sG[(1 * 32 + m) * 64 + nn];
        float o = sG[(2 * 32 + m) * 64 + nn];
        size_t idx = (size_t)m * HH + n0 + nn;
        c1nf[idx] = f * h1old[idx] + (1.f - f) * z;
        o1f[idx]  = o;
    }
}

// ---------------------------------------------------------------------------
// Attention: one block per batch. scores -> masked softmax -> context k.
// ---------------------------------------------------------------------------
__global__ __launch_bounds__(256) void k_attn(
    const float* __restrict__ c1nf, const float* __restrict__ enc,
    const unsigned char* __restrict__ mask,
    float* __restrict__ att_probs, float* __restrict__ att_vec, int t)
{
    __shared__ float sc1[HH];
    __shared__ float ssc[SS];
    __shared__ float sred[64];
    const int b = blockIdx.x, tid = threadIdx.x;
    for (int i = tid; i < HH; i += 256) sc1[i] = c1nf[(size_t)b * HH + i];
    __syncthreads();
    if (tid < SS) {
        const float* e = enc + ((size_t)b * SS + tid) * EE;
        float acc = 0.f;
        for (int h = 0; h < HH; ++h) acc += sc1[h] * e[h];
        ssc[tid] = mask[(size_t)b * SS + tid] ? acc : -INFINITY;
    }
    __syncthreads();
    if (tid < 64) sred[tid] = fmaxf(ssc[tid], ssc[tid + 64]);
    __syncthreads();
    for (int off = 32; off >= 1; off >>= 1) {
        if (tid < off) sred[tid] = fmaxf(sred[tid], sred[tid + off]);
        __syncthreads();
    }
    float mx = sred[0];
    __syncthreads();
    if (tid < SS) ssc[tid] = expf(ssc[tid] - mx);
    __syncthreads();
    if (tid < 64) sred[tid] = ssc[tid] + ssc[tid + 64];
    __syncthreads();
    for (int off = 32; off >= 1; off >>= 1) {
        if (tid < off) sred[tid] += sred[tid + off];
        __syncthreads();
    }
    float inv = 1.f / sred[0];
    __syncthreads();
    if (tid < SS) {
        float a = ssc[tid] * inv;
        ssc[tid] = a;
        att_probs[((size_t)b * TT + t) * SS + tid] = a;
    }
    __syncthreads();
    for (int h = tid; h < HH; h += 256) {
        float acc = 0.f;
        for (int s = 0; s < SS; ++s)
            acc += ssc[s] * enc[((size_t)b * SS + s) * EE + h];
        att_vec[((size_t)b * TT + t) * HH + h] = acc;
    }
}

// ---------------------------------------------------------------------------
// Projection: h1n = (k.kpW^T + c1n.cpW^T + kpb + cpb) * o1 ; write state+hseq
// ---------------------------------------------------------------------------
__global__ __launch_bounds__(128) void k_proj(
    const float* __restrict__ att_vec, const float* __restrict__ c1nf,
    const __bf16* __restrict__ kpbf, const __bf16* __restrict__ cpbf,
    const float* kpb, const float* cpb,
    const float* __restrict__ o1f, float* __restrict__ h1,
    __bf16* __restrict__ hseqbf, int t)
{
    __shared__ __bf16 sA[2][32 * LDK];
    const int tid = threadIdx.x, wv = tid >> 5, lane = tid & 31;
    const int n0 = blockIdx.x * 64;
    v8f acc[2];
    acc[0] = {}; acc[1] = {};

    for (int kc = 0; kc < 16; ++kc) {
        __syncthreads();
        for (int l = tid; l < 256; l += 128) {   // stage k & c1n tiles (cvt)
            int mtx = l >> 7, rem = l & 127, r = rem >> 2, c8 = (rem & 3) * 8;
            const float* src = mtx
                ? &c1nf[(size_t)r * HH + kc * 32 + c8]
                : &att_vec[((size_t)r * TT + t) * HH + kc * 32 + c8];
            *reinterpret_cast<v8bf*>(&sA[mtx][r * LDK + c8]) = cvt8(src);
        }
        __syncthreads();
        const int nsub = n0 + wv * 16;
        v16bf aK0 = ld_frag(sA[0], LDK);
        v16bf aK1 = ld_frag(sA[0] + 16 * LDK, LDK);
        v16bf aC0 = ld_frag(sA[1], LDK);
        v16bf aC1 = ld_frag(sA[1] + 16 * LDK, LDK);
        v16bf bK = ld_frag(kpbf + (size_t)nsub * HH + kc * 32, HH);
        v16bf bC = ld_frag(cpbf + (size_t)nsub * HH + kc * 32, HH);
        acc[0] = wmma_bf16(aK0, bK, acc[0]);
        acc[0] = wmma_bf16(aC0, bC, acc[0]);
        acc[1] = wmma_bf16(aK1, bK, acc[1]);
        acc[1] = wmma_bf16(aC1, bC, acc[1]);
    }
    const int nl = lane & 15, mh = (lane & 16) ? 8 : 0;
    for (int mt = 0; mt < 2; ++mt) {
        v8f a = acc[mt];
        for (int i = 0; i < 8; ++i) {
            int m = mt * 16 + mh + i;
            int nn = n0 + wv * 16 + nl;
            float v = a[i] + kpb[nn] + cpb[nn];
            v *= o1f[(size_t)m * HH + nn];
            h1[(size_t)m * HH + nn] = v;
            hseqbf[((size_t)m * TT + t) * HH + nn] = (__bf16)v;
        }
    }
}

// ---------------------------------------------------------------------------
// Output GEMM: C[2048,32000] = hseq_bf16[2048,512] . outW[32000,512]^T
// Double-buffered pipeline: async LDS-DMA (ASYNCcnt) fills A(kc+1) and VALU
// cvt-stages B(kc+1) while WMMAs consume tile kc.
// ---------------------------------------------------------------------------
__global__ __launch_bounds__(256) void k_outgemm(
    const __bf16* __restrict__ A, const float* __restrict__ Wout,
    float* __restrict__ Cout)
{
    __shared__ __bf16 sA[2][128 * LDK];
    __shared__ __bf16 sB[2][128 * LDK];
    const int tid = threadIdx.x, wv = tid >> 5, lane = tid & 31;
    const int m0 = blockIdx.y * 128, n0 = blockIdx.x * 128;
    const int mw = wv & 3, nw = wv >> 2;
    v8f acc[2][4];
    for (int a = 0; a < 2; ++a)
        for (int b = 0; b < 4; ++b) acc[a][b] = {};

    // prologue: fill buffer 0 with chunk 0
    for (int q = tid; q < 512; q += 256) {          // A: 512 x 16B async DMA
        int r = q >> 2, c16 = q & 3;
        async_ld_b128(lds_off(&sA[0][r * LDK + c16 * 8]),
                      &A[(size_t)(m0 + r) * HH + c16 * 8]);
    }
    for (int l = tid; l < 512; l += 256) {          // B: fp32->bf16 cvt stage
        int r = l >> 2, c8 = (l & 3) * 8;
        *reinterpret_cast<v8bf*>(&sB[0][r * LDK + c8]) =
            cvt8(&Wout[(size_t)(n0 + r) * HH + c8]);
    }
    wait_async0();
    __syncthreads();

    for (int kc = 0; kc < 16; ++kc) {
        const int p = kc & 1, q2 = p ^ 1;
        // issue next tile's fills first so DMA overlaps the WMMAs below
        if (kc < 15) {
            for (int q = tid; q < 512; q += 256) {
                int r = q >> 2, c16 = q & 3;
                async_ld_b128(lds_off(&sA[q2][r * LDK + c16 * 8]),
                              &A[(size_t)(m0 + r) * HH + (kc + 1) * 32 + c16 * 8]);
            }
            for (int l = tid; l < 512; l += 256) {
                int r = l >> 2, c8 = (l & 3) * 8;
                *reinterpret_cast<v8bf*>(&sB[q2][r * LDK + c8]) =
                    cvt8(&Wout[(size_t)(n0 + r) * HH + (kc + 1) * 32 + c8]);
            }
        }
        // compute on current tile
        const __bf16* ab = sA[p] + (mw * 32) * LDK;
        const __bf16* bb = sB[p] + (nw * 64) * LDK;
        v16bf a0 = ld_frag(ab, LDK);
        v16bf a1 = ld_frag(ab + 16 * LDK, LDK);
        for (int nt = 0; nt < 4; ++nt) {
            v16bf bfr = ld_frag(bb + nt * 16 * LDK, LDK);
            acc[0][nt] = wmma_bf16(a0, bfr, acc[0][nt]);
            acc[1][nt] = wmma_bf16(a1, bfr, acc[1][nt]);
        }
        wait_async0();
        __syncthreads();
    }

    const int nl = lane & 15, mh = (lane & 16) ? 8 : 0;
    for (int mt = 0; mt < 2; ++mt)
        for (int nt = 0; nt < 4; ++nt) {
            v8f a = acc[mt][nt];
            for (int i = 0; i < 8; ++i) {
                int m = m0 + mw * 32 + mt * 16 + mh + i;
                int n = n0 + nw * 64 + nt * 16 + nl;
                Cout[(size_t)m * VV + n] = a[i];
            }
        }
}

// ---------------------------------------------------------------------------
extern "C" void kernel_launch(void* const* d_in, const int* in_sizes, int n_in,
                              void* d_out, int out_size, void* d_ws, size_t ws_size,
                              hipStream_t stream)
{
    const float* embed = (const float*)d_in[0];
    const float* enc   = (const float*)d_in[1];
    const float* ehid  = (const float*)d_in[2];
    const unsigned char* mask = (const unsigned char*)d_in[3];
    const float* zW0 = (const float*)d_in[5];
    const float* zb0 = (const float*)d_in[6];
    const float* fW0 = (const float*)d_in[7];
    const float* fb0 = (const float*)d_in[8];
    const float* oW0 = (const float*)d_in[9];
    const float* ob0 = (const float*)d_in[10];
    const float* iW0 = (const float*)d_in[11];
    const float* ib0 = (const float*)d_in[12];
    const float* epzW0 = (const float*)d_in[13];
    const float* epzb0 = (const float*)d_in[14];
    const float* epoW0 = (const float*)d_in[15];
    const float* epob0 = (const float*)d_in[16];
    const float* epiW0 = (const float*)d_in[17];
    const float* epib0 = (const float*)d_in[18];
    const float* zW1 = (const float*)d_in[19];
    const float* zb1 = (const float*)d_in[20];
    const float* fW1 = (const float*)d_in[21];
    const float* fb1 = (const float*)d_in[22];
    const float* oW1 = (const float*)d_in[23];
    const float* ob1 = (const float*)d_in[24];
    const float* epzW1 = (const float*)d_in[25];
    const float* epzb1 = (const float*)d_in[26];
    const float* epoW1 = (const float*)d_in[27];
    const float* epob1 = (const float*)d_in[28];
    const float* kpW = (const float*)d_in[29];
    const float* kpb = (const float*)d_in[30];
    const float* cpW = (const float*)d_in[31];
    const float* cpb = (const float*)d_in[32];
    const float* outW = (const float*)d_in[33];

    float* out       = (float*)d_out;
    float* att_probs = out + (size_t)BB * TT * VV;
    float* att_vec   = att_probs + (size_t)BB * TT * SS;

    // workspace carve-up (~11 MB)
    float*  ep     = (float*)d_ws;                 // 5*32*512 f32
    float*  c0     = ep + 5 * BB * HH;             // 32*512 f32 each
    float*  h1     = c0 + BB * HH;
    float*  c1nf   = h1 + BB * HH;
    float*  o1f    = c1nf + BB * HH;
    __bf16* h0bf   = (__bf16*)(o1f + BB * HH);     // 2 slots * 32*512 bf16
    __bf16* hseqbf = h0bf + 2 * BB * HH;           // 2048*512 bf16
    __bf16* w0bf   = hseqbf + (size_t)BB * TT * HH;        // 4*2*512*512
    __bf16* w1bf   = w0bf + (size_t)4 * 2 * HH * 512;      // 3*2*512*512
    __bf16* kpbf   = w1bf + (size_t)3 * 2 * HH * 512;      // 512*512
    __bf16* cpbf   = kpbf + (size_t)HH * 512;              // 512*512

    // one-time weight repack to bf16 (reused 64x by the recurrence)
    {
        const int gB = (512 * 1024 + 255) / 256;
        k_repack_tap<<<gB, 256, 0, stream>>>(zW0, w0bf + 0 * 2 * HH * 512);
        k_repack_tap<<<gB, 256, 0, stream>>>(fW0, w0bf + 1 * 2 * HH * 512);
        k_repack_tap<<<gB, 256, 0, stream>>>(oW0, w0bf + 2 * 2 * HH * 512);
        k_repack_tap<<<gB, 256, 0, stream>>>(iW0, w0bf + 3 * 2 * HH * 512);
        k_repack_tap<<<gB, 256, 0, stream>>>(zW1, w1bf + 0 * 2 * HH * 512);
        k_repack_tap<<<gB, 256, 0, stream>>>(fW1, w1bf + 1 * 2 * HH * 512);
        k_repack_tap<<<gB, 256, 0, stream>>>(oW1, w1bf + 2 * 2 * HH * 512);
        const int cB = (HH * 512 + 255) / 256;
        k_cvt<<<cB, 256, 0, stream>>>(kpW, kpbf, HH * 512);
        k_cvt<<<cB, 256, 0, stream>>>(cpW, cpbf, HH * 512);
    }

    // precompute encoder projections
    k_prep_ep<<<(5 * BB * HH + 255) / 256, 256, 0, stream>>>(
        ehid, epzW0, epzb0, epoW0, epob0, epiW0, epib0,
        epzW1, epzb1, epoW1, epob1, ep);

    // zero recurrent state: c0, h1, c1n, o1 (f32) + both h0 bf16 slots
    {
        int nwords = (4 * BB * HH * 4 + 2 * BB * HH * 2) / 4;
        k_zero<<<(nwords + 255) / 256, 256, 0, stream>>>((unsigned int*)c0, nwords);
    }

    for (int t = 0; t < TT; ++t) {
        __bf16* h0cur  = h0bf + (t & 1) * BB * HH;
        __bf16* h0prev = h0bf + ((t & 1) ^ 1) * BB * HH;
        k_layer0<<<8, 128, 0, stream>>>(embed, t, w0bf, zb0, fb0, ob0, ib0,
                                        ep, c0, h0cur);
        k_layer1<<<8, 128, 0, stream>>>(h0prev, h0cur, w1bf, zb1, fb1, ob1,
                                        ep, h1, c1nf, o1f);
        k_attn<<<32, 256, 0, stream>>>(c1nf, enc, mask, att_probs, att_vec, t);
        k_proj<<<8, 128, 0, stream>>>(att_vec, c1nf, kpbf, cpbf, kpb, cpb,
                                      o1f, h1, hseqbf, t);
    }

    // output GEMM: 2048 x 32000 x 512
    k_outgemm<<<dim3(VV / 128, (BB * TT) / 128), 256, 0, stream>>>(
        hseqbf, outW, out);
}
